// SwinTransformerBlock_6021544149105
// MI455X (gfx1250) — compile-verified, matrix operand order checked by reading, not run on hardware
//
#include <hip/hip_runtime.h>
#include <cstdint>
#include <cstddef>

// ---------------- CDNA5 WMMA types ----------------
typedef __attribute__((ext_vector_type(16))) __bf16 v16bf;
typedef __attribute__((ext_vector_type(8)))  float  v8f;

// ---------------- problem constants ----------------
#define DIMC   192
#define HEADS  6
#define HD     32
#define NTOK   343
#define NPAD   352                 // 22 tiles of 16
#define NWIN   128                 // B * 64 windows
#define ROWSP  (NWIN * NPAD)       // 45056 padded window-token rows
#define MTOK   43904               // B * 28^3 tokens (divisible by 16)
#define NN     (NTOK * NTOK)       // 117649
#define HID    768
#define QKSCALE 0.17677669529663687f   // HEAD_DIM^-0.5

// ---------------- helpers ----------------
static __device__ __forceinline__ unsigned short f2bf(float f) {
  union { float f; unsigned u; } a; a.f = f;
  unsigned r = a.u + 0x7fffu + ((a.u >> 16) & 1u);   // round-to-nearest-even
  return (unsigned short)(r >> 16);
}

static __device__ __forceinline__ v8f wmma_bf16(v16bf a, v16bf b, v8f c) {
  // D(16x16,f32) = A(16x32,bf16) * B(32x16,bf16) + C
  return __builtin_amdgcn_wmma_f32_16x16x32_bf16(false, a, false, b, (short)0, c,
                                                 false, false);
}

// 16x32 bf16 operand fragment.  CDNA5 16-bit A/B layout: lane%16 selects the
// row (A: M, B: N), element e maps to k = (lane>=16?8:0) + (e>=8?16:0) + e%8.
// Rows are k-contiguous in memory, so the fragment is two b128 loads per lane.
static __device__ __forceinline__ v16bf ldfrag(const unsigned short* p0, int stride,
                                               int lane) {
  const unsigned short* p = p0 + (size_t)(lane & 15) * stride + ((lane & 16) ? 8 : 0);
  union { uint4 u[2]; v16bf v; } r;
  r.u[0] = *reinterpret_cast<const uint4*>(p);
  r.u[1] = *reinterpret_cast<const uint4*>(p + 16);
  return r.v;
}

// Same layout but source is f32 (LDS softmax buffer); convert on the fly.
static __device__ __forceinline__ v16bf ldfrag_f32(const float* p0, int stride, int lane) {
  const float* p = p0 + (size_t)(lane & 15) * stride + ((lane & 16) ? 8 : 0);
  union { unsigned short s[16]; v16bf v; } r;
#pragma unroll
  for (int e = 0; e < 8; ++e) { r.s[e] = f2bf(p[e]); r.s[8 + e] = f2bf(p[16 + e]); }
  return r.v;
}

static __device__ __forceinline__ float wsum(float v) {
#pragma unroll
  for (int o = 16; o > 0; o >>= 1) v += __shfl_xor(v, o, 32);
  return v;
}
static __device__ __forceinline__ float wmaxr(float v) {
#pragma unroll
  for (int o = 16; o > 0; o >>= 1) v = fmaxf(v, __shfl_xor(v, o, 32));
  return v;
}

// ---------------- small prep kernels ----------------
__global__ void k_cvt_bf16(const float* __restrict__ s, unsigned short* __restrict__ d,
                           int n) {
  int i = blockIdx.x * 256 + threadIdx.x;
  if (i < n) d[i] = f2bf(s[i]);
}

__global__ void k_bias(const int* __restrict__ ridx, const float* __restrict__ tab,
                       float* __restrict__ bias) {
  int i = blockIdx.x * 256 + threadIdx.x;
  if (i >= NN) return;
  int t = ridx[i];
#pragma unroll
  for (int h = 0; h < HEADS; ++h) bias[h * NN + i] = tab[t * HEADS + h];
}

// LN1 + cyclic shift(-3) + window partition -> bf16 A matrix [ROWSP][192]
__global__ void k_ln1_part(const float* __restrict__ x, const float* __restrict__ g,
                           const float* __restrict__ bb, unsigned short* __restrict__ Aq) {
  int wave = threadIdx.x >> 5, lane = threadIdx.x & 31;
  int gw = blockIdx.x * 8 + wave;
  if (gw >= ROWSP) return;
  int win = gw / NPAD, row = gw - win * NPAD;
  unsigned short* dst = Aq + (size_t)gw * DIMC;
  if (row >= NTOK) {
#pragma unroll
    for (int j = 0; j < 6; ++j) dst[lane + 32 * j] = 0;   // zero padding rows
    return;
  }
  int bi = win >> 6, wi = win & 63;
  int db = wi >> 4, hb = (wi >> 2) & 3, wb = wi & 3;
  int td = row / 49, th = (row / 7) % 7, tw = row % 7;
  int od = (db * 7 + td + 3) % 28, oh = (hb * 7 + th + 3) % 28, ow = (wb * 7 + tw + 3) % 28;
  const float* src = x + ((size_t)bi * 21952 + (od * 784 + oh * 28 + ow)) * DIMC;
  float v[6]; float s = 0.f;
#pragma unroll
  for (int j = 0; j < 6; ++j) { v[j] = src[lane + 32 * j]; s += v[j]; }
  float mean = wsum(s) * (1.f / DIMC);
  float vs = 0.f;
#pragma unroll
  for (int j = 0; j < 6; ++j) { float d = v[j] - mean; vs += d * d; }
  float rstd = rsqrtf(wsum(vs) * (1.f / DIMC) + 1e-5f);
#pragma unroll
  for (int j = 0; j < 6; ++j) {
    int c = lane + 32 * j;
    dst[c] = f2bf((v[j] - mean) * rstd * g[c] + bb[c]);
  }
}

// ---------------- QKV GEMM: [ROWSP,192] x [192,576]^T, split epilogue ----------------
__global__ void k_qkv(const unsigned short* __restrict__ A,
                      const unsigned short* __restrict__ W,   // (576,192) bf16
                      const float* __restrict__ bias,
                      unsigned short* __restrict__ qb, unsigned short* __restrict__ kb,
                      unsigned short* __restrict__ vT) {
  int wave = threadIdx.x >> 5, lane = threadIdx.x & 31;
  int gw = blockIdx.x * 8 + wave;
  const int NT = 9, MT = ROWSP / 16;
  if (gw >= MT * NT) return;
  int mt = gw / NT, nt = gw - mt * NT;
  const unsigned short* Ab = A + (size_t)mt * 16 * DIMC;
  v8f acc[4] = {};
#pragma unroll
  for (int kk = 0; kk < 6; ++kk) {
    v16bf a = ldfrag(Ab + kk * 32, DIMC, lane);
#pragma unroll
    for (int f = 0; f < 4; ++f) {
      v16bf b = ldfrag(W + (size_t)(nt * 64 + f * 16) * DIMC + kk * 32, DIMC, lane);
      acc[f] = wmma_bf16(a, b, acc[f]);
    }
  }
  int half = (lane & 16) ? 8 : 0;
  int nl = lane & 15;
#pragma unroll
  for (int f = 0; f < 4; ++f) {
#pragma unroll
    for (int r = 0; r < 8; ++r) {
      int n = nt * 64 + f * 16 + nl;
      int rowg = mt * 16 + r + half;
      float val = acc[f][r] + bias[n];
      int which = n / DIMC, hc = n % DIMC;
      int head = hc >> 5, d = hc & 31;
      int win = rowg / NPAD, row = rowg - win * NPAD;
      int wh = win * HEADS + head;
      if (which == 0)      qb[((size_t)wh * NPAD + row) * HD + d] = f2bf(val * QKSCALE);
      else if (which == 1) kb[((size_t)wh * NPAD + row) * HD + d] = f2bf(val);
      else                 vT[((size_t)wh * HD + d) * NPAD + row] = f2bf(val);
    }
  }
}

// ---------------- attention: one block per (window, head), 2 waves ----------------
__global__ void k_attn(const unsigned short* __restrict__ qb,
                       const unsigned short* __restrict__ kb,
                       const unsigned short* __restrict__ vT,
                       const float* __restrict__ bias_attn,
                       const float* __restrict__ mask,
                       unsigned short* __restrict__ attn_out) {
  __shared__ float Sb[2][16 * NPAD];                    // 45056 B
  int wave = threadIdx.x >> 5, lane = threadIdx.x & 31;
  int wh = blockIdx.x;                                   // win*6+head
  int win = wh / HEADS, head = wh - win * HEADS, wi = win & 63;
  float* S = Sb[wave];
  const unsigned short* Q = qb + (size_t)wh * NPAD * HD;
  const unsigned short* K = kb + (size_t)wh * NPAD * HD;
  const unsigned short* V = vT + (size_t)wh * HD * NPAD;
  const float* bias = bias_attn + (size_t)head * NN;
  const float* msk  = mask + (size_t)wi * NN;
  int half = (lane & 16) ? 8 : 0;
  int nl = lane & 15;

  for (int it = wave; it < NPAD / 16; it += 2) {
    v16bf a = ldfrag(Q + (size_t)it * 16 * HD, HD, lane);   // K=32 == HEAD_DIM
#pragma unroll 1
    for (int j = 0; j < NPAD / 16; ++j) {
      v16bf b = ldfrag(K + (size_t)j * 16 * HD, HD, lane);
      v8f c = {};
      c = wmma_bf16(a, b, c);
      int mk = j * 16 + nl;
      int mk_c = mk < NTOK ? mk : NTOK - 1;
#pragma unroll
      for (int r = 0; r < 8; ++r) {
        int m = r + half;
        int nq = it * 16 + m;
        int nq_c = nq < NTOK ? nq : NTOK - 1;
        float val = c[r] + bias[nq_c * NTOK + mk_c] + msk[nq_c * NTOK + mk_c];
        if (mk >= NTOK) val = -1e30f;                   // mask padded key tokens
        S[m * NPAD + mk] = val;
      }
    }
    // row softmax (whole wave per row; same-wave LDS, no barrier needed)
    for (int m = 0; m < 16; ++m) {
      float* row = S + m * NPAD;
      float mx = -1e30f;
      for (int cix = lane; cix < NPAD; cix += 32) mx = fmaxf(mx, row[cix]);
      mx = wmaxr(mx);
      float sm = 0.f;
      for (int cix = lane; cix < NPAD; cix += 32) {
        float e = __expf(row[cix] - mx); row[cix] = e; sm += e;
      }
      sm = wsum(sm);
      float inv = 1.f / sm;
      for (int cix = lane; cix < NPAD; cix += 32) row[cix] *= inv;
    }
    // O(16x32) = P(16x352) @ V(352x32)
    v8f o0 = {}, o1 = {};
#pragma unroll 1
    for (int kk = 0; kk < NPAD / 32; ++kk) {
      v16bf a2 = ldfrag_f32(S + kk * 32, NPAD, lane);
      v16bf b0 = ldfrag(V + 0 * 16 * NPAD + kk * 32, NPAD, lane);
      v16bf b1 = ldfrag(V + 16 * NPAD + kk * 32, NPAD, lane);
      o0 = wmma_bf16(a2, b0, o0);
      o1 = wmma_bf16(a2, b1, o1);
    }
#pragma unroll
    for (int r = 0; r < 8; ++r) {
      size_t rowg = (size_t)win * NPAD + it * 16 + r + half;
      attn_out[rowg * DIMC + head * HD + nl]      = f2bf(o0[r]);
      attn_out[rowg * DIMC + head * HD + 16 + nl] = f2bf(o1[r]);
    }
  }
}

// ---------------- proj GEMM: [ROWSP,192] x [192,192]^T -> f32 ----------------
__global__ void k_proj(const unsigned short* __restrict__ A,
                       const unsigned short* __restrict__ W,
                       const float* __restrict__ bias, float* __restrict__ out) {
  int wave = threadIdx.x >> 5, lane = threadIdx.x & 31;
  int gw = blockIdx.x * 8 + wave;
  const int NT = 3, MT = ROWSP / 16;
  if (gw >= MT * NT) return;
  int mt = gw / NT, nt = gw - mt * NT;
  const unsigned short* Ab = A + (size_t)mt * 16 * DIMC;
  v8f acc[4] = {};
#pragma unroll
  for (int kk = 0; kk < 6; ++kk) {
    v16bf a = ldfrag(Ab + kk * 32, DIMC, lane);
#pragma unroll
    for (int f = 0; f < 4; ++f) {
      v16bf b = ldfrag(W + (size_t)(nt * 64 + f * 16) * DIMC + kk * 32, DIMC, lane);
      acc[f] = wmma_bf16(a, b, acc[f]);
    }
  }
  int half = (lane & 16) ? 8 : 0;
  int nl = lane & 15;
#pragma unroll
  for (int f = 0; f < 4; ++f) {
#pragma unroll
    for (int r = 0; r < 8; ++r) {
      int n = nt * 64 + f * 16 + nl;
      size_t rowg = (size_t)mt * 16 + r + half;
      out[rowg * DIMC + n] = acc[f][r] + bias[n];
    }
  }
}

// -------- un-shift gather + residual + LN2 -> x_new f32, yln bf16 --------
__global__ void k_scatter_ln2(const float* __restrict__ x, const float* __restrict__ proj,
                              const float* __restrict__ g, const float* __restrict__ bb,
                              float* __restrict__ xn, unsigned short* __restrict__ yln) {
  int wave = threadIdx.x >> 5, lane = threadIdx.x & 31;
  int gw = blockIdx.x * 8 + wave;
  if (gw >= MTOK) return;
  int bi = gw / 21952, r = gw - bi * 21952;
  int od = r / 784, oh = (r / 28) % 28, ow = r % 28;
  int sd = (od + 25) % 28, sh = (oh + 25) % 28, sw = (ow + 25) % 28;  // roll(+3)^-1
  int wi = ((sd / 7) * 4 + sh / 7) * 4 + sw / 7;
  int t  = (sd % 7) * 49 + (sh % 7) * 7 + (sw % 7);
  size_t rowg = (size_t)(bi * 64 + wi) * NPAD + t;
  float v[6]; float s = 0.f;
#pragma unroll
  for (int j = 0; j < 6; ++j) {
    int c = lane + 32 * j;
    v[j] = x[(size_t)gw * DIMC + c] + proj[rowg * DIMC + c];
    xn[(size_t)gw * DIMC + c] = v[j];
    s += v[j];
  }
  float mean = wsum(s) * (1.f / DIMC);
  float vs = 0.f;
#pragma unroll
  for (int j = 0; j < 6; ++j) { float d = v[j] - mean; vs += d * d; }
  float rstd = rsqrtf(wsum(vs) * (1.f / DIMC) + 1e-5f);
#pragma unroll
  for (int j = 0; j < 6; ++j) {
    int c = lane + 32 * j;
    yln[(size_t)gw * DIMC + c] = f2bf((v[j] - mean) * rstd * g[c] + bb[c]);
  }
}

// ---------------- fused MLP: fc1+GELU (LDS) then fc2+residual ----------------
__global__ void k_mlp(const unsigned short* __restrict__ yln,
                      const unsigned short* __restrict__ w1, const float* __restrict__ b1,
                      const unsigned short* __restrict__ w2, const float* __restrict__ b2,
                      const float* __restrict__ xn, float* __restrict__ out) {
  __shared__ __align__(16) unsigned short h1s[16 * HID];   // 24576 B
  int wave = threadIdx.x >> 5, lane = threadIdx.x & 31;
  int mt = blockIdx.x;
  const unsigned short* Ab = yln + (size_t)mt * 16 * DIMC;
  int half = (lane & 16) ? 8 : 0;
  int nl = lane & 15;
  // stage 1: h1 = gelu(yln @ fc1_w^T + b1), 4 waves cover 768 cols
#pragma unroll 1
  for (int ch = 0; ch < 3; ++ch) {
    v8f acc[4] = {};
#pragma unroll
    for (int kk = 0; kk < 6; ++kk) {
      v16bf a = ldfrag(Ab + kk * 32, DIMC, lane);
#pragma unroll
      for (int f = 0; f < 4; ++f) {
        int n0 = wave * 192 + ch * 64 + f * 16;
        v16bf b = ldfrag(w1 + (size_t)n0 * DIMC + kk * 32, DIMC, lane);
        acc[f] = wmma_bf16(a, b, acc[f]);
      }
    }
#pragma unroll
    for (int f = 0; f < 4; ++f) {
      int n = wave * 192 + ch * 64 + f * 16 + nl;
#pragma unroll
      for (int r = 0; r < 8; ++r) {
        float v = acc[f][r] + b1[n];
        v = 0.5f * v * (1.f + erff(v * 0.70710678118654752f));   // exact GELU
        h1s[(r + half) * HID + n] = f2bf(v);
      }
    }
  }
  __syncthreads();
  // stage 2: out = h1 @ fc2_w^T + b2 + x_new, 4 waves x 48 cols
  v8f acc2[3] = {};
#pragma unroll 1
  for (int kk = 0; kk < HID / 32; ++kk) {
    v16bf a = ldfrag(h1s + kk * 32, HID, lane);
#pragma unroll
    for (int f = 0; f < 3; ++f) {
      int n0 = wave * 48 + f * 16;
      v16bf b = ldfrag(w2 + (size_t)n0 * HID + kk * 32, HID, lane);
      acc2[f] = wmma_bf16(a, b, acc2[f]);
    }
  }
#pragma unroll
  for (int f = 0; f < 3; ++f) {
    int n = wave * 48 + f * 16 + nl;
#pragma unroll
    for (int r = 0; r < 8; ++r) {
      size_t row = (size_t)mt * 16 + r + half;
      out[row * DIMC + n] = acc2[f][r] + b2[n] + xn[row * DIMC + n];
    }
  }
}

// ---------------- launcher ----------------
extern "C" void kernel_launch(void* const* d_in, const int* in_sizes, int n_in,
                              void* d_out, int out_size, void* d_ws, size_t ws_size,
                              hipStream_t stream) {
  (void)in_sizes; (void)n_in; (void)out_size; (void)ws_size;
  const float* x        = (const float*)d_in[0];
  const float* mask     = (const float*)d_in[1];
  const int*   rel_idx  = (const int*)d_in[2];
  const float* rel_tab  = (const float*)d_in[3];
  const float* n1g      = (const float*)d_in[4];
  const float* n1b      = (const float*)d_in[5];
  const float* qkv_w    = (const float*)d_in[6];
  const float* qkv_b    = (const float*)d_in[7];
  const float* proj_w   = (const float*)d_in[8];
  const float* proj_b   = (const float*)d_in[9];
  const float* n2g      = (const float*)d_in[10];
  const float* n2b      = (const float*)d_in[11];
  const float* fc1_w    = (const float*)d_in[12];
  const float* fc1_b    = (const float*)d_in[13];
  const float* fc2_w    = (const float*)d_in[14];
  const float* fc2_b    = (const float*)d_in[15];
  float* out = (float*)d_out;

  // workspace layout (bytes); regions re-used by stage lifetime (~107 MB total)
  char* ws = (char*)d_ws;
  unsigned short* Aq       = (unsigned short*)(ws + 0);          // 17,301,504
  unsigned short* attn_out = Aq;                                  // alias (A dead)
  unsigned short* qb       = (unsigned short*)(ws + 17301504);    // 17,301,504
  unsigned short* yln      = qb;                                  // alias (q dead)
  unsigned short* kb       = (unsigned short*)(ws + 34603008);    // 17,301,504
  float*          proj_out = (float*)(ws + 34603008);             // alias k+vT (34,603,008)
  unsigned short* vT       = (unsigned short*)(ws + 51904512);    // 17,301,504
  float*          bias_att = (float*)(ws + 69206016);             // 2,823,576
  float*          xnew     = (float*)(ws + 72032256);             // 33,718,272
  unsigned short* wq       = (unsigned short*)(ws + 105750528);   // 221,184
  unsigned short* wp       = (unsigned short*)(ws + 105971712);   // 73,728
  unsigned short* w1       = (unsigned short*)(ws + 106045440);   // 294,912
  unsigned short* w2       = (unsigned short*)(ws + 106340352);   // 294,912  (end 106,635,264)

  k_cvt_bf16<<<(110592 + 255) / 256, 256, 0, stream>>>(qkv_w, wq, 110592);
  k_cvt_bf16<<<(36864 + 255) / 256, 256, 0, stream>>>(proj_w, wp, 36864);
  k_cvt_bf16<<<(147456 + 255) / 256, 256, 0, stream>>>(fc1_w, w1, 147456);
  k_cvt_bf16<<<(147456 + 255) / 256, 256, 0, stream>>>(fc2_w, w2, 147456);
  k_bias<<<(NN + 255) / 256, 256, 0, stream>>>(rel_idx, rel_tab, bias_att);

  k_ln1_part<<<ROWSP / 8, 256, 0, stream>>>(x, n1g, n1b, Aq);
  k_qkv<<<(ROWSP / 16) * 9 / 8, 256, 0, stream>>>(Aq, wq, qkv_b, qb, kb, vT);
  k_attn<<<NWIN * HEADS, 64, 0, stream>>>(qb, kb, vT, bias_att, mask, attn_out);
  k_proj<<<(ROWSP / 16) * 3 / 8, 256, 0, stream>>>(attn_out, wp, proj_b, proj_out);
  k_scatter_ln2<<<MTOK / 8, 256, 0, stream>>>(x, proj_out, n2g, n2b, xnew, yln);
  k_mlp<<<MTOK / 16, 128, 0, stream>>>(yln, w1, fc1_b, w2, fc2_b, xnew, out);
}
